// MambaBlock_6631429505698
// MI455X (gfx1250) — matrix operand order repaired and checked by reading, not compiled, not run on hardware
//
#include <hip/hip_runtime.h>
#include <hip/hip_bf16.h>

typedef __attribute__((ext_vector_type(16))) __bf16 v16bf;
typedef __attribute__((ext_vector_type(8)))  __bf16 v8bf;
typedef __attribute__((ext_vector_type(8)))  float  v8f;

#define DIM      2048
#define DINNER   4096
#define LSEQ     1024
#define DTRANK   128
#define NSTATE   16
#define XDBL_LD  160   // dt_rank + 2*NSTATE

// ---------------------------------------------------------------------------
// f32 -> bf16 bulk convert (8 elems / thread, vector loads + 16B store)
// ---------------------------------------------------------------------------
__global__ void k_cvt_bf16(const float* __restrict__ s, __bf16* __restrict__ d) {
    const size_t i = ((size_t)blockIdx.x * 256 + threadIdx.x) * 8;
    const float4 f0 = *(const float4*)(s + i);
    const float4 f1 = *(const float4*)(s + i + 4);
    v8bf r;
    r[0] = (__bf16)f0.x; r[1] = (__bf16)f0.y; r[2] = (__bf16)f0.z; r[3] = (__bf16)f0.w;
    r[4] = (__bf16)f1.x; r[5] = (__bf16)f1.y; r[6] = (__bf16)f1.z; r[7] = (__bf16)f1.w;
    *(v8bf*)(d + i) = r;
}

// dtr (x_dbl[:, :128]) -> packed bf16 (1024 x 128)
__global__ void k_cvt_dtr(const float* __restrict__ xdbl, __bf16* __restrict__ dtrb) {
    const int i = blockIdx.x * 256 + threadIdx.x;   // over LSEQ*DTRANK
    const int l = i >> 7;
    const int c = i & (DTRANK - 1);
    dtrb[i] = (__bf16)xdbl[l * XDBL_LD + c];
}

// ---------------------------------------------------------------------------
// Fused residual add + LayerNorm. One block (256 thr) per row of 2048.
// Writes res (second output region, f32) and normalized h (bf16 for WMMA).
// ---------------------------------------------------------------------------
__global__ void k_resnorm(const float* __restrict__ hs, const float* __restrict__ rs,
                          const float* __restrict__ w,  const float* __restrict__ b,
                          __bf16* __restrict__ hout, float* __restrict__ resout) {
    const int l   = blockIdx.x;
    const int tid = threadIdx.x;
    __shared__ float red[256];

    float v[8];
    float s = 0.f;
#pragma unroll
    for (int j = 0; j < 8; ++j) {
        const int c = tid + j * 256;
        const float r = hs[l * DIM + c] + rs[l * DIM + c];
        v[j] = r;
        resout[l * DIM + c] = r;
        s += r;
    }
    red[tid] = s; __syncthreads();
    for (int off = 128; off > 0; off >>= 1) {
        if (tid < off) red[tid] += red[tid + off];
        __syncthreads();
    }
    const float mu = red[0] * (1.f / DIM);
    __syncthreads();

    float q = 0.f;
#pragma unroll
    for (int j = 0; j < 8; ++j) { const float d = v[j] - mu; q += d * d; }
    red[tid] = q; __syncthreads();
    for (int off = 128; off > 0; off >>= 1) {
        if (tid < off) red[tid] += red[tid + off];
        __syncthreads();
    }
    const float rstd = rsqrtf(red[0] * (1.f / DIM) + 1e-5f);
#pragma unroll
    for (int j = 0; j < 8; ++j) {
        const int c = tid + j * 256;
        hout[l * DIM + c] = (__bf16)((v[j] - mu) * rstd * w[c] + b[c]);
    }
}

// ---------------------------------------------------------------------------
// bf16 WMMA GEMM:  C[M,N] = A[M,K] * W[N,K]^T   (bf16 in, f32 accum/out)
// 256-thread blocks = 8 waves on one WGP. All 8 waves use the SAME weight
// (B) fragments but different M rows, so the weight stream is fetched from
// L2 once per block and re-served to the other 7 waves by the WGP$.
// Per wave: TM x TN 16x16 C tiles; hot loop = global_load_b128 -> v_wmma.
// Fragment layouts per CDNA5 ISA 16-bit A (16x32) / B (32x16) tables:
//   A lane: M = lane&15; elems 0..7 = K (8*half)+0..7, elems 8..15 = +16
//   B lane: N = lane&15; elems 0..7 = K (8*half)+0..7, elems 8..15 = +16
//   C lane: elem r -> (M = 8*half + r, N = lane&15)
// ---------------------------------------------------------------------------
__device__ __forceinline__ v16bf cat8(v8bf lo, v8bf hi) {
    return __builtin_shufflevector(lo, hi, 0,1,2,3,4,5,6,7,8,9,10,11,12,13,14,15);
}

template <int TM, int TN>
__global__ void k_gemm(const __bf16* __restrict__ A, int lda,
                       const __bf16* __restrict__ W, int ldw,
                       float* __restrict__ C, int ldc, int K) {
    const int lane = threadIdx.x & 31;
    const int wave = threadIdx.x >> 5;               // 0..7
    const int half = lane >> 4;
    const int lrow = lane & 15;
    const int m0 = (blockIdx.y * 8 + wave) * (16 * TM);  // 8 waves stack along M
    const int n0 = blockIdx.x * (16 * TN);               // shared N across waves

    const __bf16* arow[TM];
#pragma unroll
    for (int m = 0; m < TM; ++m)
        arow[m] = A + (size_t)(m0 + 16 * m + lrow) * lda + 8 * half;
    const __bf16* wrow[TN];
#pragma unroll
    for (int t = 0; t < TN; ++t)
        wrow[t] = W + (size_t)(n0 + 16 * t + lrow) * ldw + 8 * half;

    v8f acc[TM][TN];
#pragma unroll
    for (int m = 0; m < TM; ++m)
#pragma unroll
        for (int t = 0; t < TN; ++t)
            acc[m][t] = (v8f){0.f,0.f,0.f,0.f,0.f,0.f,0.f,0.f};

#pragma unroll 2
    for (int k0 = 0; k0 < K; k0 += 32) {
        v16bf afrag[TM];
#pragma unroll
        for (int m = 0; m < TM; ++m) {
            __builtin_prefetch(arow[m] + k0 + 64, 0, 3);
            afrag[m] = cat8(*(const v8bf*)(arow[m] + k0),
                            *(const v8bf*)(arow[m] + k0 + 16));
        }
#pragma unroll
        for (int t = 0; t < TN; ++t) {
            const v16bf bfrag = cat8(*(const v8bf*)(wrow[t] + k0),
                                     *(const v8bf*)(wrow[t] + k0 + 16));
#pragma unroll
            for (int m = 0; m < TM; ++m)
                acc[m][t] = __builtin_amdgcn_wmma_f32_16x16x32_bf16(
                                false, afrag[m], false, bfrag, (short)0,
                                acc[m][t], false, false);
        }
    }
#pragma unroll
    for (int m = 0; m < TM; ++m)
#pragma unroll
        for (int t = 0; t < TN; ++t) {
            float* cp = C + (size_t)(m0 + 16 * m + 8 * half) * ldc + n0 + t * 16 + lrow;
#pragma unroll
            for (int r = 0; r < 8; ++r) cp[(size_t)r * ldc] = acc[m][t][r];
        }
}

// ---------------------------------------------------------------------------
// Causal depthwise conv (width 4) + bias + SiLU over x = xz[:, :DINNER].
// Emits f32 (for the scan) and bf16 (for the x-proj GEMM).
// ---------------------------------------------------------------------------
__global__ void k_conv_silu(const float* __restrict__ xz, const float* __restrict__ cw,
                            const float* __restrict__ cb, float* __restrict__ xc,
                            __bf16* __restrict__ xcb) {
    const int i = blockIdx.x * 256 + threadIdx.x;   // over LSEQ*DINNER
    const int d = i & (DINNER - 1);
    const int l = i >> 12;
    float acc = cb[d];
#pragma unroll
    for (int k = 0; k < 4; ++k) {
        const int lt = l + k - 3;
        if (lt >= 0) acc += xz[(size_t)lt * (2 * DINNER) + d] * cw[d * 4 + k];
    }
    acc = acc / (1.f + __expf(-acc));                // silu
    xc[i]  = acc;
    xcb[i] = (__bf16)acc;
}

// ---------------------------------------------------------------------------
// dt = softplus(dt_pre + b_dt)
// ---------------------------------------------------------------------------
__global__ void k_softplus(float* __restrict__ dt, const float* __restrict__ bdt) {
    const int i = blockIdx.x * 256 + threadIdx.x;
    const int d = i & (DINNER - 1);
    const float v = dt[i] + bdt[d];
    dt[i] = (v > 20.f) ? v : log1pf(__expf(v));
}

// ---------------------------------------------------------------------------
// Selective scan. Thread = one (d, n) state; n lives in the low 4 lane bits,
// so the 16-state y-reduction is a 4-step shfl_xor butterfly within a wave.
// Fused epilogue: y = (scan + x*D) * silu(z), stored bf16 for the out GEMM.
// ---------------------------------------------------------------------------
__global__ void k_scan(const float* __restrict__ dt, const float* __restrict__ xc,
                       const float* __restrict__ xdbl, const float* __restrict__ xz,
                       const float* __restrict__ A_log, const float* __restrict__ Dp,
                       __bf16* __restrict__ y) {
    const int g = blockIdx.x * blockDim.x + threadIdx.x;   // 65536
    const int d = g >> 4;
    const int n = g & (NSTATE - 1);
    const float negA = -__expf(A_log[d * NSTATE + n]);
    const float Dd = Dp[d];
    float st = 0.f;

    for (int t = 0; t < LSEQ; ++t) {
        const float dtv = dt[(size_t)t * DINNER + d];
        const float xv  = xc[(size_t)t * DINNER + d];
        const float Bn  = xdbl[(size_t)t * XDBL_LD + DTRANK + n];
        const float Cn  = xdbl[(size_t)t * XDBL_LD + DTRANK + NSTATE + n];

        st = st * __expf(dtv * negA) + (dtv * xv) * Bn;
        float p = st * Cn;
        p += __shfl_xor(p, 1, 32);
        p += __shfl_xor(p, 2, 32);
        p += __shfl_xor(p, 4, 32);
        p += __shfl_xor(p, 8, 32);
        if (n == 0) {
            const float zv = xz[(size_t)t * (2 * DINNER) + DINNER + d];
            y[(size_t)t * DINNER + d] =
                (__bf16)((p + xv * Dd) * (zv / (1.f + __expf(-zv))));
        }
    }
}

// ---------------------------------------------------------------------------
extern "C" void kernel_launch(void* const* d_in, const int* in_sizes, int n_in,
                              void* d_out, int out_size, void* d_ws, size_t ws_size,
                              hipStream_t stream) {
    const float* hs   = (const float*)d_in[0];   // hidden_states (1,1024,2048)
    const float* rsd  = (const float*)d_in[1];   // residual
    const float* nw   = (const float*)d_in[2];   // norm_w
    const float* nb   = (const float*)d_in[3];   // norm_b
    const float* Win  = (const float*)d_in[4];   // (8192,2048)
    const float* cw   = (const float*)d_in[5];   // (4096,4)
    const float* cb   = (const float*)d_in[6];   // (4096,)
    const float* Wx   = (const float*)d_in[7];   // (160,4096)
    const float* Wdt  = (const float*)d_in[8];   // (4096,128)
    const float* bdt  = (const float*)d_in[9];   // (4096,)
    const float* Alog = (const float*)d_in[10];  // (4096,16)
    const float* Dp   = (const float*)d_in[11];  // (4096,)
    const float* Wout = (const float*)d_in[12];  // (2048,4096)

    float* out = (float*)d_out;                  // (1024,2048)
    float* res = out + (size_t)LSEQ * DIM;       // second output region

    const size_t MB = 1u << 20;
    char* ws = (char*)d_ws;
    float*  xzb   = (float*)(ws);                 // 32 MiB : xz (1024,8192) f32
    float*  xcb   = (float*)(ws + 32 * MB);       // 16 MiB : conv(x) f32
    float*  dtb   = (float*)(ws + 48 * MB);       // 16 MiB : dt f32
    float*  xdbl  = (float*)(ws + 64 * MB);       //  1 MiB : x_dbl (1024,160) f32
    __bf16* hb    = (__bf16*)(ws + 65 * MB);      //  4 MiB : h bf16
    __bf16* xcb16 = (__bf16*)(ws + 69 * MB);      //  8 MiB : conv(x) bf16
    __bf16* yb    = (__bf16*)(ws + 77 * MB);      //  8 MiB : y bf16
    __bf16* dtrb  = (__bf16*)(ws + 85 * MB);      // .25MiB : dtr bf16
    __bf16* Wx16  = (__bf16*)(ws + 86 * MB);      // 1.25MiB
    __bf16* Wdt16 = (__bf16*)(ws + 88 * MB);      //  1 MiB
    __bf16* Wbig  = (__bf16*)(ws + 89 * MB);      // 32 MiB : W_in bf16, then W_out bf16

    // --- weight conversions (small weights up front; W_in into shared big slot)
    k_cvt_bf16<<<(2 * DINNER * DIM) / 2048, 256, 0, stream>>>(Win, Wbig);
    k_cvt_bf16<<<(XDBL_LD * DINNER) / 2048, 256, 0, stream>>>(Wx, Wx16);
    k_cvt_bf16<<<(DINNER * DTRANK) / 2048, 256, 0, stream>>>(Wdt, Wdt16);

    // 1) residual + layernorm (res straight to d_out, h as bf16)
    k_resnorm<<<LSEQ, 256, 0, stream>>>(hs, rsd, nw, nb, hb, res);

    // 2) xz = h @ W_in^T : M=1024 N=8192 K=2048   (256-row macro-tile / block)
    k_gemm<2, 4><<<dim3(8192 / 64, LSEQ / 256), 256, 0, stream>>>(
        hb, DIM, Wbig, DIM, xzb, 2 * DINNER, DIM);

    // 3) causal conv + silu over x half (f32 + bf16 outputs)
    k_conv_silu<<<(LSEQ * DINNER) / 256, 256, 0, stream>>>(xzb, cw, cb, xcb, xcb16);

    // 4) x_dbl = x @ W_xproj^T : M=1024 N=160 K=4096
    k_gemm<2, 1><<<dim3(XDBL_LD / 16, LSEQ / 256), 256, 0, stream>>>(
        xcb16, DINNER, Wx16, DINNER, xdbl, XDBL_LD, DINNER);

    // 5) dtr -> bf16, then dt_pre = dtr @ W_dt^T : M=1024 N=4096 K=128
    k_cvt_dtr<<<(LSEQ * DTRANK) / 256, 256, 0, stream>>>(xdbl, dtrb);
    k_gemm<2, 4><<<dim3(DINNER / 64, LSEQ / 256), 256, 0, stream>>>(
        dtrb, DTRANK, Wdt16, DTRANK, dtb, DINNER, DTRANK);

    // 6) dt = softplus(dt_pre + b_dt)
    k_softplus<<<(LSEQ * DINNER) / 256, 256, 0, stream>>>(dtb, bdt);

    // 7) selective scan + gating epilogue (y as bf16)
    k_scan<<<(DINNER * NSTATE) / 256, 256, 0, stream>>>(
        dtb, xcb, xdbl, xzb, Alog, Dp, yb);

    // 8) W_out bf16 into the (now free) big slot, then out = y @ W_out^T
    k_cvt_bf16<<<(DIM * DINNER) / 2048, 256, 0, stream>>>(Wout, Wbig);
    k_gemm<2, 4><<<dim3(DIM / 64, LSEQ / 256), 256, 0, stream>>>(
        yb, DINNER, Wbig, DINNER, out, DIM, DINNER);
}
